// ScaledDotProductAttention_64768106823937
// MI455X (gfx1250) — matrix operand order repaired and checked by reading, not compiled, not run on hardware
//
#include <hip/hip_runtime.h>

// Problem constants (B,H,L,DK,DV) = (4,16,1024,64,64)
#define NHEADS 64       // B*H
#define SEQ    1024
#define DKDIM  64
#define DVDIM  64
#define WAVES  4        // waves per block, one 16-row q-tile each; all share one head
#define VPAD   40       // padded key-row length (f16) for transposed V  (80 B, 16B-aligned)
#define KPAD   72       // padded dk-row length (f16) for K             (144 B, 16B-aligned)

typedef _Float16 v16h __attribute__((ext_vector_type(16)));
typedef _Float16 v8h  __attribute__((ext_vector_type(8)));
typedef float    v8f  __attribute__((ext_vector_type(8)));
typedef float    v4f  __attribute__((ext_vector_type(4)));

// Softmax shift constant: cancels in normalization, keeps exp() in f16 range.
#define PSHIFT 8.0f

__global__ __launch_bounds__(128) void
attn_fused_kernel(const float* __restrict__ Q, const float* __restrict__ K,
                  const float* __restrict__ V, const unsigned char* __restrict__ mask,
                  float* __restrict__ ctx, float* __restrict__ attn)
{
    // Block-shared, double-buffered staged chunks (32 keys each), pre-converted to f16:
    __shared__ _Float16 vlds[2][DVDIM][VPAD];   // transposed: [dv][key]
    __shared__ _Float16 klds[2][32][KPAD];      // row-major:  [key][dk]
    // Per-wave private LDS (same-wave DS ops are in-order; no barriers needed).
    __shared__ unsigned char mtile[WAVES][16][32];  // mask tile via async-to-LDS
    __shared__ _Float16 ptile[WAVES][16][32];   // P tile (D-layout -> A-layout transpose)
    __shared__ float    red[WAVES][32][8];      // row-sum partials
    __shared__ float    sums[WAVES][16];        // final row sums

    const int tid  = threadIdx.x;
    const int w    = tid >> 5;
    const int lane = tid & 31;
    const int half = lane >> 4;       // 0: lanes 0-15, 1: lanes 16-31
    const int lr   = lane & 15;

    const int tile = blockIdx.x * WAVES + w;    // 0..4095
    const int head = tile >> 6;                 // 64 q-tiles per (b,h); uniform in block
    const int q0   = (tile & 63) * 16;

    const float*         Qh = Q    + (size_t)head * SEQ * DKDIM;
    const float*         Kh = K    + (size_t)head * SEQ * DKDIM;
    const float*         Vh = V    + (size_t)head * SEQ * DVDIM;
    const unsigned char* Mh = mask + (size_t)head * SEQ * SEQ;
    float*               Ah = attn + (size_t)head * SEQ * SEQ;
    float*               Ch = ctx  + (size_t)head * SEQ * DVDIM;

    // Staging roles of this thread.
    const int sdv  = tid & 63;        // V: one dv column ...
    const int skh  = tid >> 6;        //    ... half of the 32-key chunk
    const int skey = tid >> 2;        // K: one key row ...
    const int sdk0 = (tid & 3) * 16;  //    ... quarter of the dk dimension

    // Per-lane fixed addresses for the async mask tile (lane = one mask row slice).
    const unsigned       mlds = (unsigned)(unsigned long long)&mtile[w][lr][16 * half];
    const unsigned char* mrow = Mh + (size_t)(q0 + lr) * SEQ + 16 * half;

    // ---------------- Q tile -> two f16 A-fragments (16x32 each) ----------------
    // A layout (16-bit): lane = m (+16 for K-half), V0-3: K=8*half+0..7, V4-7: K=16+8*half+0..7
    v16h aq[2];
#pragma unroll
    for (int c = 0; c < 2; ++c) {
        const float* qrow = Qh + (size_t)(q0 + lr) * DKDIM + 32 * c;
        const v4f* lo = (const v4f*)(qrow + 8 * half);
        const v4f* hi = (const v4f*)(qrow + 16 + 8 * half);
        v4f x0 = lo[0], x1 = lo[1], y0 = hi[0], y1 = hi[1];
#pragma unroll
        for (int j = 0; j < 4; ++j) {
            aq[c][j]      = (_Float16)x0[j];
            aq[c][4 + j]  = (_Float16)x1[j];
            aq[c][8 + j]  = (_Float16)y0[j];
            aq[c][12 + j] = (_Float16)y1[j];
        }
    }

    v8f cacc[4] = {};     // context accumulators, 4 col-tiles of 16 (DV=64)
    v8f rs = {};          // per-lane row-sum partials (8 rows in D-layout)

    // ---------------- stream over key tiles, in pairs of 16 (32-key chunks) ----------------
    for (int u = 0; u < 32; ++u) {
        const int kk0 = 32 * u;
        const int buf = u & 1;

        // ---- async mask tile: 16x32 bytes in ONE instruction per wave (ASYNCcnt) ----
        {
            const unsigned long long gsrc = (unsigned long long)(mrow + kk0);
            asm volatile("global_load_async_to_lds_b128 %0, %1, off"
                         :: "v"(mlds), "v"(gsrc) : "memory");
        }

        // ---- cooperative staging: f32 global -> f16 LDS, once per block ----
        {
            // V chunk, transposed to [dv][key]
            float vt[16];
#pragma unroll
            for (int i = 0; i < 16; ++i)
                vt[i] = Vh[(size_t)(kk0 + skh * 16 + i) * DVDIM + sdv];   // coalesced b32
            v8h a, b;
#pragma unroll
            for (int i = 0; i < 8; ++i) { a[i] = (_Float16)vt[i]; b[i] = (_Float16)vt[8 + i]; }
            v8h* vdst = (v8h*)&vlds[buf][sdv][skh * 16];
            vdst[0] = a;
            vdst[1] = b;

            // K chunk, row-major [key][dk]
            const v4f* krow = (const v4f*)(Kh + (size_t)(kk0 + skey) * DKDIM + sdk0);
            v4f z0 = krow[0], z1 = krow[1], z2 = krow[2], z3 = krow[3];   // coalesced b128
            v8h ka, kb;
#pragma unroll
            for (int j = 0; j < 4; ++j) {
                ka[j]     = (_Float16)z0[j];
                ka[4 + j] = (_Float16)z1[j];
                kb[j]     = (_Float16)z2[j];
                kb[4 + j] = (_Float16)z3[j];
            }
            v8h* kdst = (v8h*)&klds[buf][skey][sdk0];
            kdst[0] = ka;
            kdst[1] = kb;

            // speculative prefetch of next chunk's lines (dropped silently if OOB)
            const int nk = (u < 31) ? kk0 + 32 : kk0;
            __builtin_prefetch(&Vh[(size_t)(nk + skh * 16) * DVDIM + sdv], 0, 1);
            __builtin_prefetch(&Kh[(size_t)(nk + skey) * DKDIM + sdk0], 0, 1);
        }
        __syncthreads();   // staged buffers ready; buf last consumed before barrier u-1

        // drain this wave's async mask tile before use
        asm volatile("s_wait_asynccnt 0x0" ::: "memory");

        // ---- two 16-key score tiles ----
#pragma unroll
        for (int pt = 0; pt < 2; ++pt) {
            const int n0 = kk0 + pt * 16;

            // K tile B-fragments straight from LDS: lane = key, dk = 8*half+0..7 and 16+8*half+0..7
            v16h bk[2];
#pragma unroll
            for (int c = 0; c < 2; ++c) {
                const v8h* kp = (const v8h*)&klds[buf][pt * 16 + lr][32 * c];
                v8h b0 = kp[half], b1 = kp[2 + half];
#pragma unroll
                for (int j = 0; j < 8; ++j) { bk[c][j] = b0[j]; bk[c][8 + j] = b1[j]; }
            }

            // S tile = Q * K^T  (fp32 accumulate)
            v8f s = {};
            s = __builtin_amdgcn_wmma_f32_16x16x32_f16(false, aq[0], false, bk[0],
                                                       (short)0, s, false, false);
            s = __builtin_amdgcn_wmma_f32_16x16x32_f16(false, aq[1], false, bk[1],
                                                       (short)0, s, false, false);

            // scale (1/sqrt(H)=0.25 per reference's d_k bug), mask, exp; stream unnormalized
            // probs to global attn, f16 copy to per-wave LDS for the PV GEMM.
#pragma unroll
            for (int r = 0; r < 8; ++r) {
                const int m = r + 8 * half;                    // D-layout: row
                const size_t idx = (size_t)(q0 + m) * SEQ + n0 + lr;
                float val = s[r] * 0.25f;
                val = mtile[w][m][16 * pt + lr] ? -1.0e9f : val;
                float p = __expf(val - PSHIFT);                // shift cancels at normalize
                rs[r] += p;
                Ah[idx] = p;
                ptile[w][m][16 * pt + lr] = (_Float16)p;
            }
        }
        __builtin_amdgcn_wave_barrier();   // pin per-wave LDS store->load ordering

        // P chunk (16x32) from LDS as A-fragment
        v16h ap;
        {
            const _Float16* prow = &ptile[w][lr][0];
            v8h lo = *(const v8h*)(prow + 8 * half);
            v8h hi = *(const v8h*)(prow + 16 + 8 * half);
#pragma unroll
            for (int j = 0; j < 8; ++j) { ap[j] = lo[j]; ap[8 + j] = hi[j]; }
        }

        // context += P * V over this 32-key chunk; B-fragments from staged LDS
#pragma unroll
        for (int t = 0; t < 4; ++t) {
            const v8h* vp = (const v8h*)&vlds[buf][16 * t + lr][0];
            v8h b0 = vp[half], b1 = vp[2 + half];
            v16h bv;
#pragma unroll
            for (int j = 0; j < 8; ++j) { bv[j] = b0[j]; bv[8 + j] = b1[j]; }
            cacc[t] = __builtin_amdgcn_wmma_f32_16x16x32_f16(false, ap, false, bv,
                                                             (short)0, cacc[t], false, false);
        }
    }

    // ---------------- row-sum reduction across the 16 lanes of each half ----------------
#pragma unroll
    for (int r = 0; r < 8; ++r) red[w][lane][r] = rs[r];
    __builtin_amdgcn_wave_barrier();
    if (lane < 16) {
        const int base = (lane >> 3) * 16;   // which lane-half holds this row
        const int r = lane & 7;
        float ssum = 0.0f;
#pragma unroll
        for (int j = 0; j < 16; ++j) ssum += red[w][base + j][r];
        sums[w][lane] = ssum;
    }
    __builtin_amdgcn_wave_barrier();

    // ---------------- context: scale by 1/rowsum and store ----------------
    float cinv[8];
#pragma unroll
    for (int r = 0; r < 8; ++r) cinv[r] = 1.0f / sums[w][r + 8 * half];
#pragma unroll
    for (int t = 0; t < 4; ++t)
#pragma unroll
        for (int r = 0; r < 8; ++r)
            Ch[(size_t)(q0 + r + 8 * half) * DVDIM + 16 * t + lr] = cacc[t][r] * cinv[r];

    // ---------------- normalize attn rows in place (L2-hot round trip) ----------------
    for (int m = 0; m < 16; ++m) {
        const float inv = 1.0f / sums[w][m];
        v4f* row = (v4f*)(Ah + (size_t)(q0 + m) * SEQ);
#pragma unroll
        for (int i = 0; i < 8; ++i) {
            v4f v = row[lane + 32 * i];
            v *= inv;
            row[lane + 32 * i] = v;
        }
    }
}

extern "C" void kernel_launch(void* const* d_in, const int* in_sizes, int n_in,
                              void* d_out, int out_size, void* d_ws, size_t ws_size,
                              hipStream_t stream) {
    (void)in_sizes; (void)n_in; (void)d_ws; (void)ws_size; (void)out_size;
    const float*         Q    = (const float*)d_in[0];
    const float*         K    = (const float*)d_in[1];
    const float*         V    = (const float*)d_in[2];
    const unsigned char* mask = (const unsigned char*)d_in[3];  // jnp.bool_ -> 1 byte

    // tuple output: context [4,16,1024,64] then attn [4,16,1024,1024], concatenated flat
    float* ctx  = (float*)d_out;
    float* attn = (float*)d_out + (size_t)NHEADS * SEQ * DVDIM;

    // 4096 q-tiles, 4 waves (tiles) per 128-thread block, one head per block
    attn_fused_kernel<<<dim3(4096 / WAVES), dim3(32 * WAVES), 0, stream>>>(
        Q, K, V, mask, ctx, attn);
}
